// CustomGPTNeoXLayer_66099546685813
// MI455X (gfx1250) — matrix operand order repaired and checked by reading, not compiled, not run on hardware
//
#include <hip/hip_runtime.h>
#include <cstdint>
#include <cstddef>

// ---------------------------------------------------------------------------
// GPT-NeoX layer for MI455X (gfx1250): bf16 WMMA everywhere.
// B=2 S=2048 H=16 HD=128 HID=2048 FF=8192 ROT=32
// ---------------------------------------------------------------------------

#define BB   2
#define SS   2048
#define NH   16
#define HDIM 128
#define HID  2048
#define FF   8192
#define NTOK (BB * SS)       // 4096
#define NQKV (3 * HID)       // 6144
#define ROT  32

typedef __attribute__((ext_vector_type(16))) __bf16 v16bf;
typedef __attribute__((ext_vector_type(8)))  float  v8f;

union FragB16 { v16bf v; unsigned short s[16]; uint4 q[2]; };
union U8x16   { uint4 u; unsigned short s[8]; };

__device__ __forceinline__ unsigned short f2bf(float f) {
  union { float f; uint32_t u; } c; c.f = f;
  uint32_t u = c.u;
  uint32_t r = u + 0x7FFFu + ((u >> 16) & 1u);   // round-nearest-even
  return (unsigned short)(r >> 16);
}
__device__ __forceinline__ float bf2f(unsigned short h) {
  union { uint32_t u; float f; } c; c.u = ((uint32_t)h) << 16;
  return c.f;
}

__device__ __forceinline__ v8f wmma_bf16(const v16bf a, const v16bf b, v8f c) {
  // D = A(16x32 bf16) * B(32x16 bf16) + C(16x16 f32)
  return __builtin_amdgcn_wmma_f32_16x16x32_bf16(
      false, a, false, b, (short)0, c, false, false);
}

// ---------------------------------------------------------------------------
// CDNA5 async global->LDS copies (ASYNCcnt-tracked), with portable fallback.
// Builtin takes (AS1 int4* src, AS3 int4* dst, imm offset, imm cpol); the
// LDS aperture keeps the byte offset in flat addr[31:0] (ISA sec 10.2), so
// the AS(3) pointer is the low 32 bits of the flat pointer.
// ---------------------------------------------------------------------------
#if defined(__has_builtin)
#if __has_builtin(__builtin_amdgcn_global_load_async_to_lds_b128) && \
    __has_builtin(__builtin_amdgcn_s_wait_asynccnt)
#define USE_ASYNC_LDS 1
#endif
#endif

#ifdef USE_ASYNC_LDS
typedef int v4i_t __attribute__((vector_size(16)));
typedef __attribute__((address_space(1))) v4i_t as1_v4i;
typedef __attribute__((address_space(3))) v4i_t as3_v4i;
__device__ __forceinline__ void async_cp16(const unsigned short* g,
                                           unsigned short* l) {
  __builtin_amdgcn_global_load_async_to_lds_b128(
      (as1_v4i*)(uintptr_t)g, (as3_v4i*)(uint32_t)(uintptr_t)l, 0, 0);
}
__device__ __forceinline__ void async_wait0() {
  __builtin_amdgcn_s_wait_asynccnt(0);
}
#else
__device__ __forceinline__ void async_cp16(const unsigned short* g,
                                           unsigned short* l) {
  *(uint4*)l = *(const uint4*)g;
}
__device__ __forceinline__ void async_wait0() {}
#endif

// ---------------------------------------------------------------------------
// Kernel 0: weight convert+transpose, W(f32, KxN) -> Wt(bf16, NxK).
// 64x64 tiles via LDS; grid (N/64, K/64), block 256.
// ---------------------------------------------------------------------------
__global__ __launch_bounds__(256) void wconv_kernel(
    const float* __restrict__ W, unsigned short* __restrict__ Wt,
    int K, int N) {
  __shared__ unsigned short T[64][65];     // +1 pad vs bank conflicts
  const int n0 = blockIdx.x * 64;
  const int k0 = blockIdx.y * 64;
  for (int i = threadIdx.x; i < 4096; i += 256) {
    int k = i >> 6, n = i & 63;            // coalesced read along n
    T[n][k] = f2bf(W[(size_t)(k0 + k) * N + n0 + n]);
  }
  __syncthreads();
  for (int i = threadIdx.x; i < 4096; i += 256) {
    int n = i >> 6, k = i & 63;            // coalesced write along k
    Wt[(size_t)(n0 + n) * K + k0 + k] = T[n][k];
  }
}

// ---------------------------------------------------------------------------
// Kernel 1: dual LayerNorm (shared stats), f32 in -> bf16 out x2
// grid = NTOK rows, block = 256
// ---------------------------------------------------------------------------
__global__ __launch_bounds__(256) void ln_dual_kernel(
    const float* __restrict__ x,
    const float* __restrict__ g1, const float* __restrict__ b1,
    const float* __restrict__ g2, const float* __restrict__ b2,
    unsigned short* __restrict__ out1, unsigned short* __restrict__ out2) {
  int row = blockIdx.x;
  const float* xr = x + (size_t)row * HID;
  float lsum = 0.f, lsq = 0.f;
  for (int i = threadIdx.x; i < HID; i += 256) {
    float v = xr[i];
    lsum += v; lsq += v * v;
  }
  #pragma unroll
  for (int m = 16; m >= 1; m >>= 1) {
    lsum += __shfl_xor(lsum, m, 32);
    lsq  += __shfl_xor(lsq,  m, 32);
  }
  __shared__ float s_sum[8], s_sq[8];
  int wave = threadIdx.x >> 5;
  if ((threadIdx.x & 31) == 0) { s_sum[wave] = lsum; s_sq[wave] = lsq; }
  __syncthreads();
  float tsum = 0.f, tsq = 0.f;
  #pragma unroll
  for (int i = 0; i < 8; ++i) { tsum += s_sum[i]; tsq += s_sq[i]; }
  float mu  = tsum * (1.0f / HID);
  float var = tsq * (1.0f / HID) - mu * mu;
  float rs  = rsqrtf(var + 1e-5f);
  for (int i = threadIdx.x; i < HID; i += 256) {
    float nv = (xr[i] - mu) * rs;
    out1[(size_t)row * HID + i] = f2bf(nv * g1[i] + b1[i]);
    out2[(size_t)row * HID + i] = f2bf(nv * g2[i] + b2[i]);
  }
}

// ---------------------------------------------------------------------------
// Kernel 2: tiled bf16 GEMM, C = A(bf16 MxK) * Wt(bf16 NxK, pre-transposed)
// BM=BN=64 BK=32, 128 threads = 4 waves; double-buffered LDS, async copies.
// Wave w owns rows [w*16,w*16+16), all 64 cols. EPI: 0 bias, 1 +GELU, 2 +adds.
// ---------------------------------------------------------------------------
template <int EPI>
__global__ __launch_bounds__(128) void gemm_bf16_kernel(
    const unsigned short* __restrict__ A, const unsigned short* __restrict__ Wt,
    const float* __restrict__ bias,
    float* __restrict__ outf, unsigned short* __restrict__ outb,
    const float* __restrict__ add0, const float* __restrict__ add1,
    int M, int N, int K) {
  __shared__ unsigned short As[2][64 * 32];    // [m][k]
  __shared__ unsigned short Bs[2][64 * 32];    // [n][k]

  const int n0   = blockIdx.x * 64;
  const int m0   = blockIdx.y * 64;
  const int tid  = threadIdx.x;
  const int wv   = tid >> 5;
  const int lane = tid & 31;
  const int g    = lane >> 4;
  const int l15  = lane & 15;

  v8f acc[4];
  #pragma unroll
  for (int i = 0; i < 4; ++i) acc[i] = v8f{};

  auto stage = [&](int buf, int k0) {
    #pragma unroll
    for (int i = tid; i < 256; i += 128) {       // 2 x 16B per array per thread
      int e = i * 8;
      int r = e >> 5, k = e & 31;                // r: row (A) / col-row (Wt)
      async_cp16(A  + (size_t)(m0 + r) * K + k0 + k, &As[buf][e]);
      async_cp16(Wt + (size_t)(n0 + r) * K + k0 + k, &Bs[buf][e]);
    }
  };

  const int nk = K >> 5;
  stage(0, 0);
  async_wait0();
  __syncthreads();

  for (int kt = 0; kt < nk; ++kt) {
    const int cur = kt & 1;
    if (kt + 1 < nk) stage(cur ^ 1, (kt + 1) << 5);   // overlap with WMMAs

    FragB16 af;
    {
      const uint4* ap = (const uint4*)(&As[cur][(wv * 16 + l15) * 32]);
      af.q[0] = ap[g];
      af.q[1] = ap[2 + g];
    }
    #pragma unroll
    for (int ni = 0; ni < 4; ++ni) {
      FragB16 bf;
      const uint4* bp = (const uint4*)(&Bs[cur][(ni * 16 + l15) * 32]);
      bf.q[0] = bp[g];
      bf.q[1] = bp[2 + g];
      acc[ni] = wmma_bf16(af.v, bf.v, acc[ni]);
    }
    async_wait0();
    __syncthreads();
  }

  // ---- epilogue: C layout row = g*8+r, col = l15 ----
  #pragma unroll
  for (int ni = 0; ni < 4; ++ni) {
    int gcol = n0 + ni * 16 + l15;
    float bv = bias[gcol];
    #pragma unroll
    for (int r = 0; r < 8; ++r) {
      int grow = m0 + wv * 16 + g * 8 + r;
      float v = acc[ni][r] + bv;
      if (EPI == 1) {                                  // exact GELU
        v = 0.5f * v * (1.0f + erff(v * 0.70710678118654752f));
      } else if (EPI == 2) {
        size_t idx = (size_t)grow * N + gcol;
        v += add0[idx] + add1[idx];
      }
      size_t oidx = (size_t)grow * N + gcol;
      if (outb) outb[oidx] = f2bf(v);
      if (outf) outf[oidx] = v;
    }
  }
}

// ---------------------------------------------------------------------------
// Kernel 3: split qkv (bf16 token-major) -> per-head Q,K,V [B,H,S,HD] bf16,
// applying rotary to the first ROT dims of q and k.
// ---------------------------------------------------------------------------
__global__ __launch_bounds__(HDIM) void rope_split_kernel(
    const unsigned short* __restrict__ qkv,
    unsigned short* __restrict__ Q, unsigned short* __restrict__ K,
    unsigned short* __restrict__ V) {
  int tok = blockIdx.x;
  int b = tok / SS, s = tok % SS;
  int d = threadIdx.x;

  float c = 1.f, sn = 0.f;
  int nb = d;
  float sign = 1.f;
  if (d < ROT) {
    int dm = d & 15;                                   // freq index
    float inv = __powf(10000.0f, -(2.0f * (float)dm) / (float)ROT);
    float ang = (float)s * inv;
    c = __cosf(ang); sn = __sinf(ang);
    nb = d ^ 16;                                       // rotate partner
    sign = (d < 16) ? -1.0f : 1.0f;
  }

  const unsigned short* base = qkv + (size_t)tok * NQKV;
  for (int h = 0; h < NH; ++h) {
    const unsigned short* hb = base + h * (3 * HDIM);
    float qv = bf2f(hb[d]);
    float kv = bf2f(hb[HDIM + d]);
    float vv = bf2f(hb[2 * HDIM + d]);
    if (d < ROT) {
      float qr = sign * bf2f(hb[nb]);
      float kr = sign * bf2f(hb[HDIM + nb]);
      qv = qv * c + qr * sn;
      kv = kv * c + kr * sn;
    }
    size_t off = (((size_t)(b * NH + h)) * SS + s) * HDIM + d;
    Q[off] = f2bf(qv);
    K[off] = f2bf(kv);
    V[off] = f2bf(vv);
  }
}

// ---------------------------------------------------------------------------
// Kernel 4: flash attention, one wave (32 threads) per 16-query tile.
// Streams 32-key tiles: 8 score WMMAs + online softmax + 8 PV WMMAs.
// ---------------------------------------------------------------------------
__global__ __launch_bounds__(32) void attn_kernel(
    const unsigned short* __restrict__ Q, const unsigned short* __restrict__ K,
    const unsigned short* __restrict__ V, unsigned short* __restrict__ ctx) {
  const int qt = blockIdx.x;       // 0..127
  const int h  = blockIdx.y;       // 0..15
  const int b  = blockIdx.z;       // 0..1
  const int lane = threadIdx.x;
  const int g = lane >> 4, l15 = lane & 15;
  const int q0 = qt * 16;

  const size_t hoff = ((size_t)(b * NH + h)) * SS * HDIM;
  const unsigned short* Qh = Q + hoff;
  const unsigned short* Kh = K + hoff;
  const unsigned short* Vh = V + hoff;

  FragB16 qf[4];
  {
    const unsigned short* qrow = Qh + (size_t)(q0 + l15) * HDIM;
    #pragma unroll
    for (int cch = 0; cch < 4; ++cch) {
      const uint4* qp = (const uint4*)(qrow + cch * 32);
      qf[cch].q[0] = qp[g];
      qf[cch].q[1] = qp[2 + g];
    }
  }

  v8f oacc[8];
  float m_i[8], l_i[8];
  #pragma unroll
  for (int t = 0; t < 8; ++t) oacc[t] = v8f{};
  #pragma unroll
  for (int r = 0; r < 8; ++r) { m_i[r] = -3.0e38f; l_i[r] = 0.f; }

  __shared__ unsigned short Pl[16 * 32];     // P tile, [m][k] A-layout staging
  __shared__ unsigned short Vt[HDIM * 32];   // V transposed: [d][k]

  const float scale = 0.088388347648318447f;  // 1/sqrt(128)
  const int nkt = (q0 + 16 + 31) >> 5;        // causal key-tile count

  for (int kt = 0; kt < nkt; ++kt) {
    const int kb = kt * 32;

    // ---- transpose V tile into LDS (lane <-> key) ----
    {
      const unsigned short* vrow = Vh + (size_t)(kb + lane) * HDIM;
      #pragma unroll
      for (int d0 = 0; d0 < HDIM; d0 += 8) {
        U8x16 t8; t8.u = *(const uint4*)(vrow + d0);
        #pragma unroll
        for (int e = 0; e < 8; ++e) Vt[(d0 + e) * 32 + lane] = t8.s[e];
      }
    }
    __syncthreads();

    // ---- scores: two 16x16 tiles, contraction over HD=128 (4 WMMAs each)
    v8f sc[2];
    #pragma unroll
    for (int t = 0; t < 2; ++t) {
      v8f sacc = v8f{};
      const unsigned short* krow = Kh + (size_t)(kb + t * 16 + l15) * HDIM;
      #pragma unroll
      for (int cch = 0; cch < 4; ++cch) {
        FragB16 kf;
        const uint4* kp = (const uint4*)(krow + cch * 32);
        kf.q[0] = kp[g];
        kf.q[1] = kp[2 + g];
        sacc = wmma_bf16(qf[cch].v, kf.v, sacc);
      }
      sc[t] = sacc;
    }

    // ---- scale + causal mask ----
    float p[2][8];
    #pragma unroll
    for (int t = 0; t < 2; ++t)
      #pragma unroll
      for (int r = 0; r < 8; ++r) {
        float sv = sc[t][r] * scale;
        int qrow = q0 + g * 8 + r;
        int kcol = kb + t * 16 + l15;
        p[t][r] = (kcol > qrow) ? -3.0e38f : sv;
      }

    // ---- online softmax (row lives in one 16-lane half) ----
    float alpha[8];
    #pragma unroll
    for (int r = 0; r < 8; ++r) {
      float m = fmaxf(p[0][r], p[1][r]);
      #pragma unroll
      for (int msk = 1; msk < 16; msk <<= 1) m = fmaxf(m, __shfl_xor(m, msk, 32));
      float nm = fmaxf(m_i[r], m);
      alpha[r] = __expf(m_i[r] - nm);
      m_i[r] = nm;
      float e0 = __expf(p[0][r] - nm);
      float e1 = __expf(p[1][r] - nm);
      p[0][r] = e0; p[1][r] = e1;
      float sm = e0 + e1;
      #pragma unroll
      for (int msk = 1; msk < 16; msk <<= 1) sm += __shfl_xor(sm, msk, 32);
      l_i[r] = l_i[r] * alpha[r] + sm;
    }
    #pragma unroll
    for (int t = 0; t < 8; ++t)
      #pragma unroll
      for (int r = 0; r < 8; ++r) oacc[t][r] *= alpha[r];

    // ---- P: C-layout -> bf16 A-layout via LDS ----
    #pragma unroll
    for (int t = 0; t < 2; ++t)
      #pragma unroll
      for (int r = 0; r < 8; ++r)
        Pl[(g * 8 + r) * 32 + t * 16 + l15] = f2bf(p[t][r]);
    __syncthreads();

    FragB16 pf;
    {
      const uint4* pp = (const uint4*)(Pl + l15 * 32);
      pf.q[0] = pp[g];
      pf.q[1] = pp[2 + g];
    }

    // ---- ctx += P * V : 8 WMMAs over HD ----
    #pragma unroll
    for (int dt = 0; dt < 8; ++dt) {
      FragB16 vf;
      const uint4* vp = (const uint4*)(Vt + (dt * 16 + l15) * 32);
      vf.q[0] = vp[g];
      vf.q[1] = vp[2 + g];
      oacc[dt] = wmma_bf16(pf.v, vf.v, oacc[dt]);
    }
    __syncthreads();
  }

  // ---- normalize and store: ctx[b, q, h*HD + d] bf16 ----
  #pragma unroll
  for (int dt = 0; dt < 8; ++dt) {
    #pragma unroll
    for (int r = 0; r < 8; ++r) {
      float o = oacc[dt][r] / l_i[r];
      int qrow = q0 + g * 8 + r;
      int dcol = dt * 16 + l15;
      ctx[((size_t)b * SS + qrow) * HID + h * HDIM + dcol] = f2bf(o);
    }
  }
}

// ---------------------------------------------------------------------------
// launch
// ---------------------------------------------------------------------------
extern "C" void kernel_launch(void* const* d_in, const int* in_sizes, int n_in,
                              void* d_out, int out_size, void* d_ws, size_t ws_size,
                              hipStream_t stream) {
  (void)in_sizes; (void)n_in; (void)out_size; (void)ws_size;
  const float* hidden = (const float*)d_in[0];
  const float* ln1_g  = (const float*)d_in[1];
  const float* ln1_b  = (const float*)d_in[2];
  const float* ln2_g  = (const float*)d_in[3];
  const float* ln2_b  = (const float*)d_in[4];
  const float* W_qkv  = (const float*)d_in[5];
  const float* b_qkv  = (const float*)d_in[6];
  const float* W_o    = (const float*)d_in[7];
  const float* b_o    = (const float*)d_in[8];
  const float* W_fc   = (const float*)d_in[9];
  const float* b_fc   = (const float*)d_in[10];
  const float* W_proj = (const float*)d_in[11];
  const float* b_proj = (const float*)d_in[12];
  float* out = (float*)d_out;

  // workspace carve-up
  char* ws = (char*)d_ws;
  size_t p = 0;
  auto carve = [&](size_t bytes) -> char* {
    char* r = ws + p;
    p += (bytes + 255) & ~(size_t)255;
    return r;
  };
  unsigned short* xln1  = (unsigned short*)carve((size_t)NTOK * HID * 2);
  unsigned short* mln2  = (unsigned short*)carve((size_t)NTOK * HID * 2);
  unsigned short* qkvb  = (unsigned short*)carve((size_t)NTOK * NQKV * 2);
  unsigned short* Qr    = (unsigned short*)carve((size_t)BB * NH * SS * HDIM * 2);
  unsigned short* Kr    = (unsigned short*)carve((size_t)BB * NH * SS * HDIM * 2);
  unsigned short* Vr    = (unsigned short*)carve((size_t)BB * NH * SS * HDIM * 2);
  unsigned short* ctxb  = (unsigned short*)carve((size_t)NTOK * HID * 2);
  float*          attno = (float*)carve((size_t)NTOK * HID * 4);
  unsigned short* hfc   = (unsigned short*)carve((size_t)NTOK * FF * 2);
  unsigned short* wtqkv = (unsigned short*)carve((size_t)HID * NQKV * 2);
  unsigned short* wto   = (unsigned short*)carve((size_t)HID * HID * 2);
  unsigned short* wtfc  = (unsigned short*)carve((size_t)HID * FF * 2);
  unsigned short* wtprj = (unsigned short*)carve((size_t)FF * HID * 2);

  // 0) one-time weight convert+transpose -> bf16 [N][K]
  wconv_kernel<<<dim3(NQKV / 64, HID / 64), 256, 0, stream>>>(W_qkv, wtqkv, HID, NQKV);
  wconv_kernel<<<dim3(HID / 64, HID / 64), 256, 0, stream>>>(W_o, wto, HID, HID);
  wconv_kernel<<<dim3(FF / 64, HID / 64), 256, 0, stream>>>(W_fc, wtfc, HID, FF);
  wconv_kernel<<<dim3(HID / 64, FF / 64), 256, 0, stream>>>(W_proj, wtprj, FF, HID);

  // 1) dual layernorm
  ln_dual_kernel<<<NTOK, 256, 0, stream>>>(hidden, ln1_g, ln1_b, ln2_g, ln2_b,
                                           xln1, mln2);
  // 2) qkv = ln1 @ W_qkv + b_qkv  (bf16 out)
  gemm_bf16_kernel<0><<<dim3(NQKV / 64, NTOK / 64), 128, 0, stream>>>(
      xln1, wtqkv, b_qkv, nullptr, qkvb, nullptr, nullptr, NTOK, NQKV, HID);
  // 3) rope + head split
  rope_split_kernel<<<NTOK, HDIM, 0, stream>>>(qkvb, Qr, Kr, Vr);
  // 4) flash attention
  attn_kernel<<<dim3(SS / 16, NH, BB), 32, 0, stream>>>(Qr, Kr, Vr, ctxb);
  // 5) attn_out = ctx @ W_o + b_o (f32 out)
  gemm_bf16_kernel<0><<<dim3(HID / 64, NTOK / 64), 128, 0, stream>>>(
      ctxb, wto, b_o, attno, nullptr, nullptr, nullptr, NTOK, HID, HID);
  // 6) h_fc = gelu(ln2 @ W_fc + b_fc) (bf16 out)
  gemm_bf16_kernel<1><<<dim3(FF / 64, NTOK / 64), 128, 0, stream>>>(
      mln2, wtfc, b_fc, nullptr, hfc, nullptr, nullptr, NTOK, FF, HID);
  // 7) out = h_fc @ W_proj + b_proj + attn_out + hidden
  gemm_bf16_kernel<2><<<dim3(HID / 64, NTOK / 64), 128, 0, stream>>>(
      hfc, wtprj, b_proj, out, nullptr, attno, hidden, NTOK, HID, FF);
}
